// MultiHeadAttention_22084721836809
// MI455X (gfx1250) — matrix operand order repaired and checked by reading, not compiled, not run on hardware
//
#include <hip/hip_runtime.h>

typedef __attribute__((ext_vector_type(16))) _Float16 v16h;
typedef __attribute__((ext_vector_type(8)))  float    v8f;
typedef int v4i __attribute__((vector_size(4 * sizeof(int))));
typedef __attribute__((address_space(1))) v4i* gv4ip;   // global 128-bit chunk
typedef __attribute__((address_space(3))) v4i* lv4ip;   // LDS    128-bit chunk

#define B_  2
#define T_  2048
#define D_  1024
#define H_  16
#define DK_ 64

#ifndef __has_builtin
#define __has_builtin(x) 0
#endif
#if __has_builtin(__builtin_amdgcn_global_load_async_to_lds_b128)
#define HAVE_ASYNC 1
#else
#define HAVE_ASYNC 0
#endif

__device__ __forceinline__ void wait_async() {
#if HAVE_ASYNC
#if __has_builtin(__builtin_amdgcn_s_wait_asynccnt)
    __builtin_amdgcn_s_wait_asynccnt(0);
#else
    asm volatile("s_wait_asynccnt 0x0" ::: "memory");
#endif
#endif
}

// Stage a 32x64 f16 tile (4 KB) from global into LDS.
// Async path: 256 x b128 chunks, 2 per thread (128 threads), ASYNCcnt-tracked.
__device__ __forceinline__ void stage_tile(const _Float16* __restrict__ gsrc,
                                           _Float16* __restrict__ ldst, int tid) {
#if HAVE_ASYNC
#pragma unroll
    for (int t = 0; t < 2; ++t) {
        const int c = tid + t * 128;
        __builtin_amdgcn_global_load_async_to_lds_b128(
            (gv4ip)(gsrc + c * 8), (lv4ip)(ldst + c * 8), 0, 0);
    }
#else
    for (int idx = tid; idx < 32 * 64; idx += 128) ldst[idx] = gsrc[idx];
#endif
}

// ---- WMMA fragment index helpers (CDNA5 ISA 7.12.2, wave32) ----
__device__ __forceinline__ int a_kidx(int i, int hs) {
    return i + 16 * hs;
}

__device__ __forceinline__ float rmax16(float v) {
#pragma unroll
    for (int m = 1; m < 16; m <<= 1) v = fmaxf(v, __shfl_xor(v, m, 32));
    return v;
}
__device__ __forceinline__ float rsum16(float v) {
#pragma unroll
    for (int m = 1; m < 16; m <<= 1) v += __shfl_xor(v, m, 32);
    return v;
}

// ---------------------------------------------------------------------------
// Generic tiled GEMM:  out = A[4096,1024] * W[1024,1024] + bias
// MODE 0: f16 out, per-head layout [B,H,T,DK]; MODE 1: f32 out [M,N].
// 256 threads = 8 waves (4 M x 2 N), tile 64x128, K-chunk 32.
// ---------------------------------------------------------------------------
template <typename TA, int MODE>
__global__ void __launch_bounds__(256)
gemm_wmma(const TA* __restrict__ A, const float* __restrict__ W,
          const float* __restrict__ bias, void* __restrict__ outp)
{
    __shared__ _Float16 As[64 * 32];
    __shared__ _Float16 Bs[32 * 128];

    const int tid  = threadIdx.x;
    const int wave = tid >> 5, lane = tid & 31;
    const int hs = lane >> 4, ln = lane & 15;
    const int wm = wave & 3, wn = wave >> 2;
    const int m0 = blockIdx.x * 64;
    const int n0 = blockIdx.y * 128;

    v8f acc[4];
#pragma unroll
    for (int c = 0; c < 4; ++c) acc[c] = {};

    for (int k0 = 0; k0 < D_; k0 += 32) {
        __syncthreads();
        for (int idx = tid; idx < 64 * 32; idx += 256) {
            int r = idx >> 5, c = idx & 31;
            As[idx] = (_Float16)A[(size_t)(m0 + r) * D_ + k0 + c];
        }
        for (int idx = tid; idx < 32 * 128; idx += 256) {
            int r = idx >> 7, c = idx & 127;
            Bs[idx] = (_Float16)W[(size_t)(k0 + r) * D_ + n0 + c];
        }
        __syncthreads();

        v16h af;
#pragma unroll
        for (int i = 0; i < 16; ++i)
            af[i] = As[(16 * wm + ln) * 32 + a_kidx(i, hs)];
#pragma unroll
        for (int c = 0; c < 4; ++c) {
            v16h bf;
#pragma unroll
            for (int i = 0; i < 16; ++i)
                bf[i] = Bs[(16 * hs + i) * 128 + 64 * wn + 16 * c + ln];
            acc[c] = __builtin_amdgcn_wmma_f32_16x16x32_f16(
                false, af, false, bf, (short)0, acc[c], false, false);
        }
    }

#pragma unroll
    for (int c = 0; c < 4; ++c) {
        const int gn = n0 + 64 * wn + 16 * c + ln;
        const float bv = bias[gn];
#pragma unroll
        for (int r = 0; r < 8; ++r) {
            const int gm = m0 + 16 * wm + r + 8 * hs;
            const float val = acc[c][r] + bv;
            if (MODE == 0) {
                _Float16* o = (_Float16*)outp;
                const int bb = gm >> 11;
                const int t  = gm & (T_ - 1);
                const int hh = gn >> 6;
                const int dk = gn & (DK_ - 1);
                o[(size_t)((bb * H_ + hh) * T_ + t) * DK_ + dk] = (_Float16)val;
            } else {
                ((float*)outp)[(size_t)gm * D_ + gn] = val;
            }
        }
    }
}

// ---------------------------------------------------------------------------
// Fused causal flash attention: 128 threads = 4 waves x 16 query rows.
// K/V streamed in 32-key tiles, double-buffered via async global->LDS copies.
// ---------------------------------------------------------------------------
__global__ void __launch_bounds__(128)
attn_wmma(const _Float16* __restrict__ qh, const _Float16* __restrict__ kh,
          const _Float16* __restrict__ vh, _Float16* __restrict__ ctx)
{
    __shared__ _Float16 Kt[2][32 * 64];   // 8 KB double-buffered
    __shared__ _Float16 Vt[2][32 * 64];   // 8 KB
    __shared__ float    Pl[4][16 * 32];   // 8 KB per-wave P staging

    const int tid  = threadIdx.x;
    const int wave = tid >> 5, lane = tid & 31;
    const int hs = lane >> 4, ln = lane & 15;

    const int bid = blockIdx.x;
    const int qb = bid & 31;
    const int h  = (bid >> 5) & (H_ - 1);
    const int b  = bid >> 9;
    const size_t hb = (size_t)(b * H_ + h) * T_ * DK_;
    const int q0 = qb * 64 + wave * 16;

    v16h a0, a1;
#pragma unroll
    for (int i = 0; i < 16; ++i) {
        a0[i] = qh[hb + (size_t)(q0 + ln) * DK_ + a_kidx(i, hs)];
        a1[i] = qh[hb + (size_t)(q0 + ln) * DK_ + 32 + a_kidx(i, hs)];
    }

    v8f o[4];
#pragma unroll
    for (int c = 0; c < 4; ++c) o[c] = {};
    float mrow[8], lrow[8];
#pragma unroll
    for (int r = 0; r < 8; ++r) { mrow[r] = -INFINITY; lrow[r] = 0.f; }

    const int nkb = (qb + 1) * 2;

    // Prologue: start tile 0 copies.
    stage_tile(kh + hb, Kt[0], tid);
    stage_tile(vh + hb, Vt[0], tid);

    for (int kbi = 0; kbi < nkb; ++kbi) {
        const int kb  = kbi * 32;
        const int cur = kbi & 1;
        const int nxt = cur ^ 1;

        wait_async();        // my async copies for buffer `cur` complete
        __syncthreads();     // all waves' copies done; `nxt` buffer free

        if (kbi + 1 < nkb) { // overlap next tile copy with this tile's compute
            stage_tile(kh + hb + (size_t)(kb + 32) * DK_, Kt[nxt], tid);
            stage_tile(vh + hb + (size_t)(kb + 32) * DK_, Vt[nxt], tid);
        }

        const bool active = (kb <= q0 + 15);
        if (active) {
            v8f s[2];
#pragma unroll
            for (int j = 0; j < 2; ++j) {
                v16h b0, b1;   // B = K^T fragments (keys kb+16j .. +15)
#pragma unroll
                for (int i = 0; i < 16; ++i) {
                    b0[i] = Kt[cur][(16 * j + ln) * 64 + 16 * hs + i];
                    b1[i] = Kt[cur][(16 * j + ln) * 64 + 32 + 16 * hs + i];
                }
                v8f z = {};
                z = __builtin_amdgcn_wmma_f32_16x16x32_f16(false, a0, false, b0, (short)0, z, false, false);
                z = __builtin_amdgcn_wmma_f32_16x16x32_f16(false, a1, false, b1, (short)0, z, false, false);
                s[j] = z;
            }
            float alpha[8];
#pragma unroll
            for (int r = 0; r < 8; ++r) {
                const int qg = q0 + r + 8 * hs;
                float s0 = s[0][r] * 0.125f;          // 1/sqrt(64)
                float s1 = s[1][r] * 0.125f;
                if (kb + ln      > qg) s0 = -1e30f;   // causal mask
                if (kb + 16 + ln > qg) s1 = -1e30f;
                const float mx = rmax16(fmaxf(s0, s1));
                const float nm = fmaxf(mrow[r], mx);
                const float al = __expf(mrow[r] - nm);
                const float p0 = __expf(s0 - nm);
                const float p1 = __expf(s1 - nm);
                lrow[r] = lrow[r] * al + rsum16(p0 + p1);
                mrow[r] = nm;
                alpha[r] = al;
                const int m = r + 8 * hs;             // C-frag -> LDS re-layout
                Pl[wave][m * 32 + ln]      = p0;
                Pl[wave][m * 32 + 16 + ln] = p1;
            }
#pragma unroll
            for (int c = 0; c < 4; ++c)
#pragma unroll
                for (int r = 0; r < 8; ++r) o[c][r] *= alpha[r];
        }
        __syncthreads();     // Pl visible; async copies to `nxt` continue in bg
        if (active) {
            v16h pa;   // P as A-fragment (16x32)
#pragma unroll
            for (int i = 0; i < 16; ++i)
                pa[i] = (_Float16)Pl[wave][ln * 32 + a_kidx(i, hs)];
#pragma unroll
            for (int c = 0; c < 4; ++c) {
                v16h bv;
#pragma unroll
                for (int i = 0; i < 16; ++i)
                    bv[i] = Vt[cur][(16 * hs + i) * 64 + 16 * c + ln];
                o[c] = __builtin_amdgcn_wmma_f32_16x16x32_f16(
                    false, pa, false, bv, (short)0, o[c], false, false);
            }
        }
    }

#pragma unroll
    for (int c = 0; c < 4; ++c)
#pragma unroll
        for (int r = 0; r < 8; ++r) {
            const int t   = q0 + r + 8 * hs;
            const int dim = h * DK_ + 16 * c + ln;
            ctx[(size_t)(b * T_ + t) * D_ + dim] = (_Float16)(o[c][r] / lrow[r]);
        }
}

// ---------------------------------------------------------------------------
extern "C" void kernel_launch(void* const* d_in, const int* in_sizes, int n_in,
                              void* d_out, int out_size, void* d_ws, size_t ws_size,
                              hipStream_t stream)
{
    const float* Q    = (const float*)d_in[0];
    const float* K    = (const float*)d_in[1];
    const float* V    = (const float*)d_in[2];
    // d_in[3] = causal mask, recomputed analytically in-kernel
    const float* Wq_w = (const float*)d_in[4];
    const float* Wq_b = (const float*)d_in[5];
    const float* Wk_w = (const float*)d_in[6];
    const float* Wk_b = (const float*)d_in[7];
    const float* Wv_w = (const float*)d_in[8];
    const float* Wv_b = (const float*)d_in[9];
    const float* Wo_w = (const float*)d_in[10];
    const float* Wo_b = (const float*)d_in[11];

    const size_t per = (size_t)B_ * T_ * D_;   // 4M f16 elems each
    _Float16* qh  = (_Float16*)d_ws;
    _Float16* kh  = qh + per;
    _Float16* vh  = kh + per;
    _Float16* ctx = vh + per;                  // total 32 MB of ws

    dim3 gg(64, 8), gb(256);
    hipLaunchKernelGGL((gemm_wmma<float, 0>), gg, gb, 0, stream, Q, Wq_w, Wq_b, (void*)qh);
    hipLaunchKernelGGL((gemm_wmma<float, 0>), gg, gb, 0, stream, K, Wk_w, Wk_b, (void*)kh);
    hipLaunchKernelGGL((gemm_wmma<float, 0>), gg, gb, 0, stream, V, Wv_w, Wv_b, (void*)vh);

    hipLaunchKernelGGL(attn_wmma, dim3(B_ * H_ * (T_ / 64)), dim3(128), 0, stream,
                       qh, kh, vh, ctx);

    hipLaunchKernelGGL((gemm_wmma<_Float16, 1>), gg, gb, 0, stream, ctx, Wo_w, Wo_b, d_out);
}
